// Head_59871844106436
// MI455X (gfx1250) — matrix operand order repaired and checked by reading, not compiled, not run on hardware
//
#include <hip/hip_runtime.h>
#include <hip/hip_bf16.h>

typedef __bf16 bf16_t;
typedef __attribute__((ext_vector_type(16))) bf16_t v16bf;
typedef __attribute__((ext_vector_type(8)))  float  v8f;
typedef __attribute__((ext_vector_type(4)))  float  v4f;
typedef __attribute__((ext_vector_type(8)))  unsigned short v8u16;

#define N_EMBD   1024
#define HEADSZ   128
#define SEQ      2048
#define NBATCH   8
#define BT       (NBATCH * SEQ)          // 16384 rows

static __device__ __forceinline__ unsigned short f2bfbits(float f) {
    union { float f; unsigned u; } a; a.f = f;
    unsigned r = a.u + 0x7FFFu + ((a.u >> 16) & 1u);   // round-to-nearest-even
    return (unsigned short)(r >> 16);
}

// XOR-swizzle within 32-lane group (stays inside each 16-lane half for mask<16)
#define SWZ(v, mask) __int_as_float(__builtin_amdgcn_ds_swizzle(__float_as_int(v), (0x1f | ((mask) << 10))))
static __device__ __forceinline__ float hmax16(float v) {
    v = fmaxf(v, SWZ(v, 1)); v = fmaxf(v, SWZ(v, 2));
    v = fmaxf(v, SWZ(v, 4)); v = fmaxf(v, SWZ(v, 8));
    return v;
}
static __device__ __forceinline__ float hsum16(float v) {
    v += SWZ(v, 1); v += SWZ(v, 2); v += SWZ(v, 4); v += SWZ(v, 8);
    return v;
}

static __device__ __forceinline__ v8f wmma_bf16(v16bf a, v16bf b, v8f c) {
    return __builtin_amdgcn_wmma_f32_16x16x32_bf16(false, a, false, b, (short)0, c, false, false);
}

// ---------------------------------------------------------------------------
// Pre-pass A: x (f32, [BT][C]) -> xb (bf16, [BT][C]); 8 elements per thread.
// ---------------------------------------------------------------------------
__global__ __launch_bounds__(256) void cvt_x_kernel(
    const float* __restrict__ x, unsigned short* __restrict__ xb)
{
    const size_t i = ((size_t)blockIdx.x * 256 + threadIdx.x) * 8;
    v4f f0 = *(const v4f*)(x + i);
    v4f f1 = *(const v4f*)(x + i + 4);
    v8u16 o;
    o[0] = f2bfbits(f0[0]); o[1] = f2bfbits(f0[1]);
    o[2] = f2bfbits(f0[2]); o[3] = f2bfbits(f0[3]);
    o[4] = f2bfbits(f1[0]); o[5] = f2bfbits(f1[1]);
    o[6] = f2bfbits(f1[2]); o[7] = f2bfbits(f1[3]);
    *(v8u16*)(xb + i) = o;
}

// ---------------------------------------------------------------------------
// Pre-pass B: Wk/Wq/Wv (f32, [C][H]) -> Wt (bf16, [3][H][C]) transposed, so a
// weight column (the WMMA B-fragment K-run) is contiguous.  m order: 0=K,1=Q,2=V.
// ---------------------------------------------------------------------------
__global__ __launch_bounds__(256) void cvt_w_kernel(
    const float* __restrict__ Wk, const float* __restrict__ Wq,
    const float* __restrict__ Wv, unsigned short* __restrict__ Wt)
{
    const int t = blockIdx.x * 256 + threadIdx.x;      // 0 .. 3*131072-1
    const int m = t >> 17;
    const int r = t & 0x1FFFF;
    const int h = r >> 10;
    const int c = r & (N_EMBD - 1);
    const float* W = (m == 0) ? Wk : (m == 1) ? Wq : Wv;
    Wt[(size_t)t] = f2bfbits(W[(size_t)c * HEADSZ + h]);   // Wt[m][h][c]
}

// ---------------------------------------------------------------------------
// Kernel 1: QKV projection, one wave per 16x16 output tile.  All fragment
// loads are contiguous b128 from pre-converted bf16 buffers.
//   Q (scaled by H^-0.5) and K stored row-major bf16 [B*T][H];
//   V stored transposed bf16 [B][H][T] so attention B-frags are contiguous.
// ---------------------------------------------------------------------------
__global__ __launch_bounds__(128) void qkv_kernel(
    const unsigned short* __restrict__ xb, const unsigned short* __restrict__ Wt,
    unsigned short* __restrict__ Qb, unsigned short* __restrict__ Kb,
    unsigned short* __restrict__ Vt)
{
    const int lane = threadIdx.x & 31;
    const int wave = threadIdx.x >> 5;
    const int tile = blockIdx.x * 4 + wave;     // 0..24575
    const int rt   = tile / 24;                 // row tile (16 rows of B*T)
    const int rem  = tile % 24;
    const int m    = rem >> 3;                  // 0=K, 1=Q, 2=V
    const int nt   = rem & 7;                   // 16-col tile of H

    const int half  = lane >> 4;
    const int l16   = lane & 15;
    const int arow  = rt * 16 + l16;            // A-matrix row for this lane
    const int koffA = half * 8;                 // A: upper lanes hold K+8
    const int koffB = half * 8;                 // B: upper lanes hold K+8
    const int bcol  = nt * 16 + l16;

    const unsigned short* ap = xb + (size_t)arow * N_EMBD + koffA;
    const unsigned short* wp = Wt + ((size_t)m * HEADSZ + bcol) * N_EMBD + koffB;

    v8f acc = {};
    for (int c = 0; c < N_EMBD; c += 32) {
        union { v16bf v; v8u16 h[2]; } a, bf;
        a.h[0]  = *(const v8u16*)(ap + c);
        a.h[1]  = *(const v8u16*)(ap + c + 16);
        bf.h[0] = *(const v8u16*)(wp + c);
        bf.h[1] = *(const v8u16*)(wp + c + 16);
        acc = wmma_bf16(a.v, bf.v, acc);
    }
    if (m == 1) {
        #pragma unroll
        for (int i = 0; i < 8; ++i) acc[i] *= 0.08838834764831845f;  // 128^-0.5
    }
    #pragma unroll
    for (int i = 0; i < 8; ++i) {
        const int g = rt * 16 + i + half * 8;   // global row in B*T
        const unsigned short v = f2bfbits(acc[i]);
        if (m == 0)      Kb[(size_t)g * HEADSZ + bcol] = v;
        else if (m == 1) Qb[(size_t)g * HEADSZ + bcol] = v;
        else {
            const int b = g >> 11, t = g & (SEQ - 1);
            Vt[((size_t)b * HEADSZ + bcol) * SEQ + t] = v;
        }
    }
}

// ---------------------------------------------------------------------------
// Kernel 2: fused causal flash attention. 4 waves/block; each wave owns 16
// query rows, streams 32-key blocks with online softmax. P goes through a
// per-wave LDS tile to convert C/D layout -> A layout (same-wave LDS is
// in-order, no workgroup barrier needed).
// ---------------------------------------------------------------------------
__global__ __launch_bounds__(128) void attn_kernel(
    const unsigned short* __restrict__ Qb, const unsigned short* __restrict__ Kb,
    const unsigned short* __restrict__ Vt, float* __restrict__ out)
{
    __shared__ unsigned short lds_p[4][16][32];

    const int lane = threadIdx.x & 31;
    const int wave = threadIdx.x >> 5;
    const int half = lane >> 4;
    const int l16  = lane & 15;

    const int b     = blockIdx.x >> 5;          // batch
    const int qb    = blockIdx.x & 31;          // 64-query block
    const int qbase = qb * 64 + wave * 16;

    const unsigned short* Q = Qb + (size_t)b * SEQ * HEADSZ;
    const unsigned short* K = Kb + (size_t)b * SEQ * HEADSZ;
    const unsigned short* V = Vt + (size_t)b * HEADSZ * SEQ;

    // Q fragments for this wave's 16 rows x 128 head dims (4 K-chunks of 32)
    v16bf qf[4];
    {
        const int qrow = qbase + l16;
        #pragma unroll
        for (int cc = 0; cc < 4; ++cc) {
            const unsigned short* qp = Q + (size_t)qrow * HEADSZ + cc * 32 + half * 8;
            union { v16bf v; v8u16 h[2]; } u;
            u.h[0] = *(const v8u16*)(qp);
            u.h[1] = *(const v8u16*)(qp + 16);
            qf[cc] = u.v;
        }
    }

    v8f acc_o[8];
    #pragma unroll
    for (int n = 0; n < 8; ++n) acc_o[n] = (v8f){};
    float mrow[8], lrow[8];
    #pragma unroll
    for (int i = 0; i < 8; ++i) { mrow[i] = -1e30f; lrow[i] = 0.0f; }

    const int nkb = qb * 2 + 2;                 // causal: keys <= qb*64+63
    for (int j = 0; j < nkb; ++j) {
        const int kb0 = j * 32;

        // S = Q * K^T for 16 queries x 32 keys (two 16x16 tiles)
        v8f s0 = {}, s1 = {};
        #pragma unroll
        for (int cc = 0; cc < 4; ++cc) {
            union { v16bf v; v8u16 h[2]; } k0, k1;
            const unsigned short* kp0 = K + (size_t)(kb0 + l16) * HEADSZ + cc * 32 + half * 8;
            k0.h[0] = *(const v8u16*)(kp0);
            k0.h[1] = *(const v8u16*)(kp0 + 16);
            const unsigned short* kp1 = K + (size_t)(kb0 + 16 + l16) * HEADSZ + cc * 32 + half * 8;
            k1.h[0] = *(const v8u16*)(kp1);
            k1.h[1] = *(const v8u16*)(kp1 + 16);
            s0 = wmma_bf16(qf[cc], k0.v, s0);
            s1 = wmma_bf16(qf[cc], k1.v, s1);
        }

        // causal mask + online softmax per row (VGPR i <-> row i + half*8)
        #pragma unroll
        for (int i = 0; i < 8; ++i) {
            const int qrow = qbase + i + half * 8;
            const float v0 = (kb0 + l16      <= qrow) ? s0[i] : -1e30f;
            const float v1 = (kb0 + 16 + l16 <= qrow) ? s1[i] : -1e30f;
            const float mb   = hmax16(fmaxf(v0, v1));
            const float mnew = fmaxf(mrow[i], mb);
            const float alpha = __expf(mrow[i] - mnew);
            mrow[i] = mnew;
            const float p0 = __expf(v0 - mnew);
            const float p1 = __expf(v1 - mnew);
            lrow[i] = alpha * lrow[i] + hsum16(p0 + p1);
            #pragma unroll
            for (int n = 0; n < 8; ++n) acc_o[n][i] *= alpha;
            lds_p[wave][i + half * 8][l16]      = f2bfbits(p0);
            lds_p[wave][i + half * 8][16 + l16] = f2bfbits(p1);
        }

        asm volatile("" ::: "memory");          // keep LDS store->load order
        __builtin_amdgcn_wave_barrier();

        // reload P in A-matrix layout (contiguous 16B rows)
        union { v16bf v; v8u16 h[2]; } pf;
        pf.h[0] = *(const v8u16*)&lds_p[wave][l16][half * 8];
        pf.h[1] = *(const v8u16*)&lds_p[wave][l16][16 + half * 8];

        // O += P * V  (V^T layout makes B-frags contiguous along keys)
        #pragma unroll
        for (int n = 0; n < 8; ++n) {
            union { v16bf v; v8u16 h[2]; } vf;
            const unsigned short* vp = V + (size_t)(n * 16 + l16) * SEQ + kb0 + half * 8;
            vf.h[0] = *(const v8u16*)(vp);
            vf.h[1] = *(const v8u16*)(vp + 16);
            acc_o[n] = wmma_bf16(pf.v, vf.v, acc_o[n]);
        }
    }

    // epilogue: divide by softmax denominator, store fp32
    #pragma unroll
    for (int i = 0; i < 8; ++i) {
        const float inv = 1.0f / lrow[i];
        const int qrow = qbase + i + half * 8;
        #pragma unroll
        for (int n = 0; n < 8; ++n)
            out[((size_t)b * SEQ + qrow) * HEADSZ + n * 16 + l16] = acc_o[n][i] * inv;
    }
}

extern "C" void kernel_launch(void* const* d_in, const int* in_sizes, int n_in,
                              void* d_out, int out_size, void* d_ws, size_t ws_size,
                              hipStream_t stream) {
    const float* x  = (const float*)d_in[0];
    const float* Wk = (const float*)d_in[1];
    const float* Wq = (const float*)d_in[2];
    const float* Wv = (const float*)d_in[3];

    // workspace layout (bf16 elements):
    //   xb : BT * C                 = 16,777,216
    //   Wt : 3 * H * C              =    393,216
    //   Qb : BT * H                 =  2,097,152
    //   Kb : BT * H
    //   Vt : B  * H * T
    unsigned short* xb = (unsigned short*)d_ws;
    unsigned short* Wt = xb + (size_t)BT * N_EMBD;
    unsigned short* Qb = Wt + (size_t)3 * HEADSZ * N_EMBD;
    unsigned short* Kb = Qb + (size_t)BT * HEADSZ;
    unsigned short* Vt = Kb + (size_t)BT * HEADSZ;

    // pre-pass conversions
    cvt_x_kernel<<<(BT * (size_t)N_EMBD) / (256 * 8), 256, 0, stream>>>(x, xb);
    cvt_w_kernel<<<(3 * HEADSZ * N_EMBD) / 256, 256, 0, stream>>>(Wk, Wq, Wv, Wt);

    // 24576 16x16 output tiles, 4 waves (tiles) per 128-thread block
    qkv_kernel<<<6144, 128, 0, stream>>>(xb, Wt, Qb, Kb, Vt);
    // 8 batches x 32 query-blocks of 64 rows
    attn_kernel<<<256, 128, 0, stream>>>(Qb, Kb, Vt, (float*)d_out);
}
